// CombinatorialCardSelector_60292750901841
// MI455X (gfx1250) — matrix operand order repaired and checked by reading, not compiled, not run on hardware
//
#include <hip/hip_runtime.h>
#include <hip/hip_bf16.h>

// ---------------------------------------------------------------------------
// CombinatorialCardSelector for MI455X (gfx1250, wave32, WMMA)
//
// All matmuls: C = A @ W^T (+bias) (+Cin). Operands are pre-converted to bf16
// (once per call for weights/cards; activations are produced in bf16 by the
// previous kernel), so the GEMM inner loop is pure b128 loads + wmma.
// Each wave computes a 32x64 strip: 2 A-frags x 4 B-frags -> 8 accumulators,
// each B fragment reused by 2 WMMAs. 4 waves (128 threads) per block.
// ---------------------------------------------------------------------------

typedef __attribute__((ext_vector_type(16))) __bf16 v16bf;
typedef __attribute__((ext_vector_type(8)))  float  v8f;
typedef unsigned short ushort_t;

#define BATCH 512
#define NCARDS 16384
#define CTXDIM 256
#define CARDDIM 512
#define HID 128
#define NSTEPS 5

__device__ __forceinline__ unsigned int f2bf_rne(float f) {
    unsigned int x = __float_as_uint(f);
    x += 0x7FFFu + ((x >> 16) & 1u);   // round-to-nearest-even
    return x >> 16;
}
__device__ __forceinline__ unsigned int pack_bf(float lo, float hi) {
    return f2bf_rne(lo) | (f2bf_rne(hi) << 16);
}

// bf16 fragment from two 8-element (16B) chunks
__device__ __forceinline__ v16bf load_frag_bf(const ushort_t* __restrict__ p0,
                                              const ushort_t* __restrict__ p1) {
    union { v16bf v; uint4 q[2]; } r;
    r.q[0] = *(const uint4*)p0;
    r.q[1] = *(const uint4*)p1;
    return r.v;
}

// ---------------------------------------------------------------------------
// C[M,N] = A[M,K]@W[N,K]^T + bias + Cin ; optional f32 and/or bf16 outputs.
// grid: (N/256, M/32), block 128 (4 waves); wave w -> cols [w*64, (w+1)*64)
// ---------------------------------------------------------------------------
__global__ __launch_bounds__(128)
void gemm_bf16_wmma(const ushort_t* __restrict__ A, const ushort_t* __restrict__ W,
                    const float* __restrict__ bias, const float* __restrict__ Cin,
                    float* __restrict__ Cf, ushort_t* __restrict__ Cbf,
                    int M, int N, int K)
{
    const int lane = threadIdx.x & 31;
    const int wave = threadIdx.x >> 5;
    const int l15  = lane & 15;
    const int half = lane >> 4;                 // 0: lanes 0-15, 1: lanes 16-31
    const int m0   = blockIdx.y * 32;
    const int n0   = blockIdx.x * 256 + wave * 64;

    union { v8f v; float f[8]; } acc[2][4];
#pragma unroll
    for (int r = 0; r < 2; ++r)
#pragma unroll
        for (int t = 0; t < 4; ++t)
#pragma unroll
            for (int v = 0; v < 8; ++v) acc[r][t].f[v] = 0.0f;

    const ushort_t* Arow0 = A + (size_t)(m0 + l15) * K;        // rows m0..m0+15
    const ushort_t* Arow1 = A + (size_t)(m0 + 16 + l15) * K;   // rows m0+16..m0+31

    for (int k0 = 0; k0 < K; k0 += 32) {
        // A fragment 16x32 bf16 (ISA layout): lanes<16 hold K{0..7,16..23},
        // lanes>=16 hold K{8..15,24..31}
        v16bf af0 = load_frag_bf(Arow0 + k0 + half * 8, Arow0 + k0 + 16 + half * 8);
        v16bf af1 = load_frag_bf(Arow1 + k0 + half * 8, Arow1 + k0 + 16 + half * 8);
#pragma unroll
        for (int t = 0; t < 4; ++t) {
            // B fragment 32x16 bf16: column n = l15, same K layout as A:
            // lanes<16 hold K{0..7,16..23}, lanes>=16 hold K{8..15,24..31}.
            // B[k][n] == W[n][k] -> contiguous along k in row-major W.
            const ushort_t* Wrow = W + (size_t)(n0 + t * 16 + l15) * K + k0 + half * 8;
            v16bf bf = load_frag_bf(Wrow, Wrow + 16);
            acc[0][t].v = __builtin_amdgcn_wmma_f32_16x16x32_bf16(
                false, af0, false, bf, (short)0, acc[0][t].v, false, false);
            acc[1][t].v = __builtin_amdgcn_wmma_f32_16x16x32_bf16(
                false, af1, false, bf, (short)0, acc[1][t].v, false, false);
        }
    }

#pragma unroll
    for (int r = 0; r < 2; ++r)
#pragma unroll
    for (int t = 0; t < 4; ++t) {
        const int n = n0 + t * 16 + l15;
        const float bb = bias ? bias[n] : 0.0f;
#pragma unroll
        for (int v = 0; v < 8; ++v) {
            const int m = m0 + r * 16 + v + half * 8;   // C/D layout: VGPR v, lane half
            const size_t off = (size_t)m * N + n;
            float val = acc[r][t].f[v] + bb;
            if (Cin) val += Cin[off];
            if (Cf)  Cf[off]  = val;
            if (Cbf) Cbf[off] = (ushort_t)f2bf_rne(val);
        }
    }
}

// ---------------------------------------------------------------------------
// f32 -> bf16 bulk convert (8 elements / thread, n % 8 == 0)
// ---------------------------------------------------------------------------
__global__ void cvt_f32_bf16(const float* __restrict__ in,
                             ushort_t* __restrict__ out, int n)
{
    const int i = (blockIdx.x * blockDim.x + threadIdx.x) * 8;
    if (i >= n) return;
    const float4* p = (const float4*)(in + i);
    float4 a = p[0], b = p[1];
    uint4 r;
    r.x = pack_bf(a.x, a.y); r.y = pack_bf(a.z, a.w);
    r.z = pack_bf(b.x, b.y); r.w = pack_bf(b.z, b.w);
    *(uint4*)(out + i) = r;
}

// ---------------------------------------------------------------------------
// LSTM elementwise: gates[B,4H] (i,f,g,o); c (f32) in place, h emitted as bf16.
// ---------------------------------------------------------------------------
__global__ void lstm_act(const float* __restrict__ gates,
                         ushort_t* __restrict__ h_bf, float* __restrict__ c)
{
    const int b = blockIdx.x;
    const int j = threadIdx.x;
    const float* g = gates + (size_t)b * 4 * HID;
    const float gi = g[j];
    const float gf = g[HID + j];
    const float gg = g[2 * HID + j];
    const float go = g[3 * HID + j];
    const float si = 1.0f / (1.0f + __expf(-gi));
    const float sf = 1.0f / (1.0f + __expf(-gf));
    const float so = 1.0f / (1.0f + __expf(-go));
    const size_t off = (size_t)b * HID + j;
    const float cc = sf * c[off] + si * tanhf(gg);
    c[off] = cc;
    h_bf[off] = (ushort_t)f2bf_rne(so * tanhf(cc));
}

// ---------------------------------------------------------------------------
// Gumbel-max sampling; updates -inf mask; gathers bf16 card row as next input.
// ---------------------------------------------------------------------------
__device__ __forceinline__ unsigned int mix_hash(unsigned int x) {
    x ^= x >> 16; x *= 0x7feb352du;
    x ^= x >> 15; x *= 0x846ca68bu;
    x ^= x >> 16;
    return x;
}

__global__ void sample_step(const float* __restrict__ logits,     // [B,N] masked
                            const ushort_t* __restrict__ card_bf, // [N,CARDDIM]
                            float* __restrict__ maskbias,         // [B,N] 0/-inf
                            ushort_t* __restrict__ xbuf_bf,       // [B,CARDDIM]
                            int* __restrict__ out_idx, int step)
{
    __shared__ float sval[256];
    __shared__ int   sidx[256];
    __shared__ int   chosen;
    const int b = blockIdx.x, tid = threadIdx.x;
    const float* row = logits + (size_t)b * NCARDS;

    float best = -INFINITY; int bi = 0;
    for (int n = tid; n < NCARDS; n += 256) {
        unsigned int h = mix_hash((unsigned int)(step * 0x9E3779B1u) ^
                                  (unsigned int)(b * 0x85EBCA77u) ^
                                  (unsigned int)n * 0xC2B2AE3Du);
        float u = ((float)h + 0.5f) * (1.0f / 4294967296.0f);
        float v = row[n] - __logf(-__logf(u));  // -inf rows stay -inf
        if (v > best) { best = v; bi = n; }
    }
    sval[tid] = best; sidx[tid] = bi;
    __syncthreads();
    for (int off = 128; off > 0; off >>= 1) {
        if (tid < off && sval[tid + off] > sval[tid]) {
            sval[tid] = sval[tid + off]; sidx[tid] = sidx[tid + off];
        }
        __syncthreads();
    }
    if (tid == 0) {
        chosen = sidx[0];
        out_idx[b] = chosen;
        maskbias[(size_t)b * NCARDS + chosen] = -INFINITY;   // future steps only
    }
    __syncthreads();
    const int idx = chosen;
    for (int k = tid; k < CARDDIM; k += 256)
        xbuf_bf[(size_t)b * CARDDIM + k] = card_bf[(size_t)idx * CARDDIM + k];
}

__global__ void zero_ws(uint4* __restrict__ p, int n16) {
    int i = blockIdx.x * blockDim.x + threadIdx.x;
    if (i < n16) p[i] = make_uint4(0u, 0u, 0u, 0u);
}

// ---------------------------------------------------------------------------
// Host-side orchestration
// ---------------------------------------------------------------------------
static inline void gemm(const ushort_t* A, const ushort_t* W, const float* bias,
                        const float* Cin, float* Cf, ushort_t* Cbf,
                        int M, int N, int K, hipStream_t s)
{
    dim3 grid(N / 256, M / 32);
    gemm_bf16_wmma<<<grid, 128, 0, s>>>(A, W, bias, Cin, Cf, Cbf, M, N, K);
}

static inline void cvt(const float* in, ushort_t* out, int n, hipStream_t s)
{
    cvt_f32_bf16<<<(n / 8 + 255) / 256, 256, 0, s>>>(in, out, n);
}

extern "C" void kernel_launch(void* const* d_in, const int* in_sizes, int n_in,
                              void* d_out, int out_size, void* d_ws, size_t ws_size,
                              hipStream_t stream)
{
    const float* ctx   = (const float*)d_in[0];   // [512, 256]
    const float* card  = (const float*)d_in[1];   // [16384, 512]
    const float* Wp    = (const float*)d_in[2];   // [512, 256]
    const float* bp    = (const float*)d_in[3];
    const float* Wih0  = (const float*)d_in[4];   // [512, 512]
    const float* Whh0  = (const float*)d_in[5];   // [512, 128]
    const float* bih0  = (const float*)d_in[6];
    const float* bhh0  = (const float*)d_in[7];
    const float* Wih1  = (const float*)d_in[8];   // [512, 128]
    const float* Whh1  = (const float*)d_in[9];   // [512, 128]
    const float* bih1  = (const float*)d_in[10];
    const float* bhh1  = (const float*)d_in[11];
    const float* Wo    = (const float*)d_in[12];  // [512, 128]
    const float* bo    = (const float*)d_in[13];
    // d_in[14] = max_selections — fixed to NSTEPS=5 per reference.

    int*   out_idx    = (int*)d_out;                       // [5, 512] int32
    float* out_logits = (float*)d_out + NSTEPS * BATCH;    // [5, 512, 16384] f32

    // ---- workspace layout ----
    // f32 region (zeroed: maskbias, gates, c0, c1)
    float* maskbias = (float*)d_ws;                         // 8388608
    float* gates    = maskbias + (size_t)BATCH * NCARDS;    // 262144
    float* c0       = gates + (size_t)BATCH * 4 * HID;      // 65536
    float* c1       = c0 + (size_t)BATCH * HID;             // 65536
    // bf16 region
    ushort_t* h0_bf   = (ushort_t*)(c1 + (size_t)BATCH * HID);  // 65536 (zeroed)
    ushort_t* h1_bf   = h0_bf + (size_t)BATCH * HID;            // 65536 (zeroed)
    ushort_t* card_bf = h1_bf + (size_t)BATCH * HID;            // 8388608
    ushort_t* Wp_bf   = card_bf + (size_t)NCARDS * CARDDIM;     // 131072
    ushort_t* Wih0_bf = Wp_bf + CARDDIM * CTXDIM;               // 262144
    ushort_t* Whh0_bf = Wih0_bf + 4 * HID * CARDDIM;            // 65536
    ushort_t* Wih1_bf = Whh0_bf + 4 * HID * HID;                // 65536
    ushort_t* Whh1_bf = Wih1_bf + 4 * HID * HID;                // 65536
    ushort_t* Wo_bf   = Whh1_bf + 4 * HID * HID;                // 65536
    ushort_t* ctx_bf  = Wo_bf + CARDDIM * HID;                  // 131072
    ushort_t* xbuf_bf = ctx_bf + (size_t)BATCH * CTXDIM;        // 262144
    ushort_t* proj_bf = xbuf_bf + (size_t)BATCH * CARDDIM;      // 262144

    // zero: maskbias + gates + c0 + c1 (f32) + h0_bf + h1_bf (bf16)
    {
        const size_t zbytes = ((size_t)BATCH * NCARDS + (size_t)BATCH * 4 * HID
                             + 2 * (size_t)BATCH * HID) * 4
                             + 2 * (size_t)BATCH * HID * 2;
        int n16 = (int)(zbytes / 16);
        zero_ws<<<(n16 + 255) / 256, 256, 0, stream>>>((uint4*)d_ws, n16);
    }

    // one-time bf16 conversions (weights, cards, context)
    cvt(card, card_bf, NCARDS * CARDDIM, stream);
    cvt(Wp,   Wp_bf,   CARDDIM * CTXDIM, stream);
    cvt(Wih0, Wih0_bf, 4 * HID * CARDDIM, stream);
    cvt(Whh0, Whh0_bf, 4 * HID * HID, stream);
    cvt(Wih1, Wih1_bf, 4 * HID * HID, stream);
    cvt(Whh1, Whh1_bf, 4 * HID * HID, stream);
    cvt(Wo,   Wo_bf,   CARDDIM * HID, stream);
    cvt(ctx,  ctx_bf,  BATCH * CTXDIM, stream);

    // x0 = ctx @ Wp^T + bp  (bf16 output only — feeds next GEMM)
    gemm(ctx_bf, Wp_bf, bp, nullptr, nullptr, xbuf_bf, BATCH, CARDDIM, CTXDIM, stream);

    // one 2-layer LSTM stack step on xbuf_bf (h bf16 / c f32 updated in place)
    auto stack_step = [&]() {
        gemm(xbuf_bf, Wih0_bf, bih0, nullptr, gates, nullptr, BATCH, 4 * HID, CARDDIM, stream);
        gemm(h0_bf,   Whh0_bf, bhh0, gates,   gates, nullptr, BATCH, 4 * HID, HID,     stream);
        lstm_act<<<BATCH, HID, 0, stream>>>(gates, h0_bf, c0);
        gemm(h0_bf,   Wih1_bf, bih1, nullptr, gates, nullptr, BATCH, 4 * HID, HID,     stream);
        gemm(h1_bf,   Whh1_bf, bhh1, gates,   gates, nullptr, BATCH, 4 * HID, HID,     stream);
        lstm_act<<<BATCH, HID, 0, stream>>>(gates, h1_bf, c1);
    };

    stack_step();   // initial step on x0

    for (int s = 0; s < NSTEPS; ++s) {
        float* logits_s = out_logits + (size_t)s * BATCH * NCARDS;
        // proj = h1 @ Wo^T + bo  (bf16 only — feeds logits GEMM)
        gemm(h1_bf, Wo_bf, bo, nullptr, nullptr, proj_bf, BATCH, CARDDIM, HID, stream);
        // logits = proj @ card^T + maskbias  (dominant GEMM, 512x16384x512)
        gemm(proj_bf, card_bf, nullptr, maskbias, logits_s, nullptr,
             BATCH, NCARDS, CARDDIM, stream);
        // Gumbel-max sample, mask update, gather next bf16 input
        sample_step<<<BATCH, 256, 0, stream>>>(logits_s, card_bf, maskbias, xbuf_bf,
                                               out_idx + (size_t)s * BATCH, s);
        stack_step();
    }
}
